// SWT_up_66030827209373
// MI455X (gfx1250) — matrix operand order repaired and checked by reading, not compiled, not run on hardware
//
#include <hip/hip_runtime.h>
#include <hip/hip_bf16.h>

// ---------------------------------------------------------------------------
// Token-Performer + 2x bilinear upsample for MI455X (gfx1250, wave32).
// GEMMs: v_wmma_f32_16x16x32_f16 fed by double-buffered
// global_load_async_to_lds_b128 (ASYNCcnt) staging.
// ---------------------------------------------------------------------------

typedef __attribute__((ext_vector_type(16))) _Float16 v16h;
typedef __attribute__((ext_vector_type(8)))  _Float16 v8h;
typedef __attribute__((ext_vector_type(8)))  float    v8f;

// ----- problem constants ----------------------------------------------------
constexpr int TOK  = 65536;   // B * T
constexpr int TPB  = 16384;   // tokens per batch (128*128)
constexpr int EMB  = 256;
constexpr int MFEA = 128;     // performer features
constexpr int KQV  = 768;     // 3*EMB
constexpr int KPAD = 64;      // DIM=48 padded to 64 for K%32==0

// ----- workspace layout (bytes) --------------------------------------------
constexpr size_t OFF_H16    = 0;                                        // [TOK,64] f16
constexpr size_t OFF_K16    = OFF_H16  + (size_t)TOK * KPAD * 2;        // [TOK,256] f16 (later: yattn16)
constexpr size_t OFF_Q16    = OFF_K16  + (size_t)TOK * EMB * 2;         // [TOK,256] f16 (later: z16)
constexpr size_t OFF_V32    = OFF_Q16  + (size_t)TOK * EMB * 2;         // [TOK,256] f32 (later: yfin32)
constexpr size_t OFF_KP16   = OFF_V32  + (size_t)TOK * EMB * 4;         // [TOK,128] f16 (later: h1 lo)
constexpr size_t OFF_QP16   = OFF_KP16 + (size_t)TOK * MFEA * 2;        // [TOK,128] f16 (later: h1 hi)
constexpr size_t OFF_Y32    = OFF_QP16 + (size_t)TOK * MFEA * 2;        // [TOK,256] f32
constexpr size_t OFF_XDK    = OFF_Y32  + (size_t)TOK * EMB * 4;         // [TOK] f32
constexpr size_t OFF_XDQ    = OFF_XDK  + (size_t)TOK * 4;
constexpr size_t OFF_DV     = OFF_XDQ  + (size_t)TOK * 4;
constexpr size_t OFF_KSUM   = OFF_DV   + (size_t)TOK * 4;               // [4,128] f32 (atomic)
constexpr size_t OFF_KPTV32 = OFF_KSUM + 4ull * MFEA * 4;               // [4,256,128] f32 (atomic)
constexpr size_t OFF_KPTV16 = OFF_KPTV32 + 4ull * EMB * MFEA * 4;       // [4,256,128] f16
constexpr size_t OFF_WKQVT  = OFF_KPTV16 + 4ull * EMB * MFEA * 2;       // [768,64] f16
constexpr size_t OFF_WPRM   = OFF_WKQVT  + (size_t)KQV * KPAD * 2;      // [128,256] f16
constexpr size_t OFF_WPROJ  = OFF_WPRM   + (size_t)MFEA * EMB * 2;      // [256,256] f16 (transposed)
constexpr size_t OFF_WM1    = OFF_WPROJ  + (size_t)EMB * EMB * 2;
constexpr size_t OFF_WM2    = OFF_WM1    + (size_t)EMB * EMB * 2;

// ----- CDNA5 async global->LDS copy (ASYNCcnt) ------------------------------
// GVS mode: mem_addr = SADDR(64) + VADDR(u32 byte offset); VDST = LDS byte addr.
__device__ inline void async_ld_b128(unsigned lds_addr, unsigned gl_off,
                                     const void* base) {
  asm volatile("global_load_async_to_lds_b128 %0, %1, %2"
               :: "v"(lds_addr), "v"(gl_off), "s"(base)
               : "memory");
}
__device__ inline void wait_async0() {
  asm volatile("s_wait_asynccnt 0" ::: "memory");
}
__device__ inline unsigned lds_off32(const void* p) {
  return (unsigned)(unsigned long long)p;   // generic LDS addr: offset in [31:0]
}

// ----- WMMA helpers ---------------------------------------------------------
__device__ inline v8f wmma_f16(v16h a, v16h b, v8f c) {
  return __builtin_amdgcn_wmma_f32_16x16x32_f16(false, a, false, b,
                                                (short)0, c, false, false);
}

// A fragment (16x32 f16, row-major tile in LDS, 16B-aligned rows, ld % 8 == 0)
// Lane L: row m = L&15, hi = L>>4; element e -> K = (e/8)*16 + hi*8 + (e%8)
__device__ inline v16h frag_ld_a(const _Float16* p, int ld) {
  const int lane = threadIdx.x & 31;
  const _Float16* r0 = p + (lane & 15) * ld + (lane >> 4) * 8;
  v8h lo = *(const v8h*)r0;
  v8h hv = *(const v8h*)(r0 + 16);
  v16h f;
#pragma unroll
  for (int i = 0; i < 8; ++i) { f[i] = lo[i]; f[i + 8] = hv[i]; }
  return f;
}

// B fragment from transposed LDS tile Bt[n][k]. Lane L: col n = L&15, hi = L>>4;
// element e -> K = (e/8)*16 + hi*8 + (e%8)
__device__ inline v16h frag_ld_b(const _Float16* p, int ld) {
  const int lane = threadIdx.x & 31;
  const _Float16* r0 = p + (lane & 15) * ld + (lane >> 4) * 8;
  v8h lo = *(const v8h*)r0;
  v8h hv = *(const v8h*)(r0 + 16);
  v16h f;
#pragma unroll
  for (int i = 0; i < 8; ++i) { f[i] = lo[i]; f[i + 8] = hv[i]; }
  return f;
}

// ----- generic token GEMM: out[T,N] = A[T,K] * Bt[N,K]^T (+ epilogue) -------
struct GP {
  const _Float16* A;   int ldA;
  const _Float16* Bt;  int ldB;
  int K; int ldOut;
  long btBatchStride;          // MODE 2: Bt advance per 16384-token batch
  const float* bias;
  const float* xd;             // MODE 1
  const float* D;              // MODE 2
  const float* skip;           // MODE 3/5
  _Float16* o16;
  _Float16* o16b;              // MODE 0: q output
  float* o32;                  // MODE 0: v output; MODE 3/5 output
};

template <int MODE>
__global__ __launch_bounds__(256) void gemm_tok(GP p) {
  // double-buffered tiles: A 128x32 (pad 48), Bt 64x32 (pad 48)
  __shared__ _Float16 As[2][128 * 48];
  __shared__ _Float16 Bs[2][64 * 48];
  const int tid  = threadIdx.x;
  const int lane = tid & 31;
  const int w    = tid >> 5;
  const int wm   = w & 3, wn = w >> 2;          // 4x2 wave grid -> 128x64 tile
  const long mBase = (long)blockIdx.y * 128;
  const int  nBase = blockIdx.x * 64;
  const _Float16* A  = p.A;
  const _Float16* Bt = p.Bt + ((MODE == 2) ? (mBase >> 14) * p.btBatchStride : 0);
  const _Float16* Abase = A + mBase * (long)p.ldA;      // + k0 per step
  const _Float16* Bbase = Bt + (long)nBase * p.ldB;

  // per-thread tile-load geometry (3 async b128 copies per thread per step)
  const int arow0 = (tid * 2) >> 2,     akc0 = ((tid * 2) & 3) * 8;
  const int arow1 = (tid * 2 + 1) >> 2, akc1 = ((tid * 2 + 1) & 3) * 8;
  const int brow  = tid >> 2,           bkc  = (tid & 3) * 8;

  auto issue = [&](int k0, int buf) {
    async_ld_b128(lds_off32(&As[buf][arow0 * 48 + akc0]),
                  (unsigned)((arow0 * (long)p.ldA + k0 + akc0) * 2), (const void*)Abase);
    async_ld_b128(lds_off32(&As[buf][arow1 * 48 + akc1]),
                  (unsigned)((arow1 * (long)p.ldA + k0 + akc1) * 2), (const void*)Abase);
    async_ld_b128(lds_off32(&Bs[buf][brow * 48 + bkc]),
                  (unsigned)((brow * (long)p.ldB + k0 + bkc) * 2), (const void*)Bbase);
  };

  const v8f vzero = {0.f, 0.f, 0.f, 0.f, 0.f, 0.f, 0.f, 0.f};
  v8f acc[2][2];
#pragma unroll
  for (int i = 0; i < 2; ++i)
#pragma unroll
    for (int j = 0; j < 2; ++j) acc[i][j] = vzero;

  issue(0, 0);
  int cur = 0;
  for (int k0 = 0; k0 < p.K; k0 += 32) {
    wait_async0();          // own copies for tile (k0) landed
    __syncthreads();        // everyone's copies landed
    if (k0 + 32 < p.K)      // stream next tile while this one computes
      issue(k0 + 32, cur ^ 1);

    v16h a0 = frag_ld_a(As[cur] + (wm * 32) * 48, 48);
    v16h a1 = frag_ld_a(As[cur] + (wm * 32 + 16) * 48, 48);
    v16h b0 = frag_ld_b(Bs[cur] + (wn * 32) * 48, 48);
    v16h b1 = frag_ld_b(Bs[cur] + (wn * 32 + 16) * 48, 48);
    acc[0][0] = wmma_f16(a0, b0, acc[0][0]);
    acc[0][1] = wmma_f16(a0, b1, acc[0][1]);
    acc[1][0] = wmma_f16(a1, b0, acc[1][0]);
    acc[1][1] = wmma_f16(a1, b1, acc[1][1]);
    __syncthreads();        // all reads of buf[cur] done before it is refilled
    cur ^= 1;
  }

#pragma unroll
  for (int i = 0; i < 2; ++i)
#pragma unroll
    for (int j = 0; j < 2; ++j)
#pragma unroll
      for (int r = 0; r < 8; ++r) {
        long t  = mBase + wm * 32 + i * 16 + (lane >> 4) * 8 + r;
        int  nG = nBase + wn * 32 + j * 16 + (lane & 15);
        float v = acc[i][j][r];
        if (MODE == 0) {                         // kqv split
          v += p.bias ? p.bias[nG] : 0.f;
          int reg = nG >> 8, col = nG & 255;
          if (reg == 0)      p.o16 [t * 256 + col] = (_Float16)v;
          else if (reg == 1) p.o16b[t * 256 + col] = (_Float16)v;
          else               p.o32 [t * 256 + col] = v;
        } else if (MODE == 1) {                  // performer features
          p.o16[t * p.ldOut + nG] =
              (_Float16)(__expf(v - p.xd[t]) * 0.08838834764831845f); // 1/sqrt(128)
        } else if (MODE == 2) {                  // attention / D
          p.o16[t * p.ldOut + nG] = (_Float16)(v / (p.D[t] + 1e-8f));
        } else if (MODE == 3) {                  // proj + v skip
          p.o32[t * p.ldOut + nG] = p.skip[t * p.ldOut + nG] + v + p.bias[nG];
        } else if (MODE == 4) {                  // mlp1 + exact gelu
          float xg = v + p.bias[nG];
          p.o16[t * p.ldOut + nG] =
              (_Float16)(0.5f * xg * (1.f + erff(xg * 0.7071067811865475f)));
        } else {                                 // MODE 5: mlp2 + y skip
          p.o32[t * p.ldOut + nG] = p.skip[t * p.ldOut + nG] + v + p.bias[nG];
        }
      }
}

// ----- kptv[b,n,m] = sum_t v[b,t,n]*kp[b,t,m]  (WMMA over T, atomic accum) --
__global__ __launch_bounds__(256) void kptv_k(const float* v32,
                                              const _Float16* kp16,
                                              float* kptv) {
  __shared__ _Float16 vT[256 * 40];   // vT[n][t_local], 80B rows (16B aligned)
  __shared__ _Float16 kpT[128 * 40];  // kpT[m][t_local]
  const int b = blockIdx.y, ch = blockIdx.x;
  const int tid = threadIdx.x, lane = tid & 31, w = tid >> 5;
  const long tb0 = (long)b * TPB + (long)ch * 512;

  const v8f vzero = {0.f, 0.f, 0.f, 0.f, 0.f, 0.f, 0.f, 0.f};
  v8f acc[2][8];
#pragma unroll
  for (int i = 0; i < 2; ++i)
#pragma unroll
    for (int m = 0; m < 8; ++m) acc[i][m] = vzero;

  const int km = tid & 127, kth = (tid >> 7) * 16;   // kp staging geometry

  for (int ks = 0; ks < 16; ++ks) {
    const long tb = tb0 + ks * 32;
    // v tile: thread owns row n=tid; pack 8 t-values -> one ds_store_b128
#pragma unroll
    for (int g = 0; g < 4; ++g) {
      v8h pk;
#pragma unroll
      for (int i = 0; i < 8; ++i)
        pk[i] = (_Float16)v32[(tb + g * 8 + i) * 256 + tid];
      *(v8h*)(vT + tid * 40 + g * 8) = pk;
    }
    // kp tile: thread owns (row m, 16 contiguous t) -> two ds_store_b128
#pragma unroll
    for (int g = 0; g < 2; ++g) {
      v8h pk;
#pragma unroll
      for (int i = 0; i < 8; ++i)
        pk[i] = kp16[(tb + kth + g * 8 + i) * 128 + km];
      *(v8h*)(kpT + km * 40 + kth + g * 8) = pk;
    }
    __syncthreads();
    v16h a0 = frag_ld_a(vT + (w * 32) * 40, 40);       // rows = n of kptv
    v16h a1 = frag_ld_a(vT + (w * 32 + 16) * 40, 40);
#pragma unroll
    for (int mt = 0; mt < 8; ++mt) {
      v16h bf = frag_ld_b(kpT + mt * 16 * 40, 40);
      acc[0][mt] = wmma_f16(a0, bf, acc[0][mt]);
      acc[1][mt] = wmma_f16(a1, bf, acc[1][mt]);
    }
    __syncthreads();
  }
  float* base = kptv + (long)b * EMB * MFEA;
#pragma unroll
  for (int i = 0; i < 2; ++i)
#pragma unroll
    for (int mt = 0; mt < 8; ++mt)
#pragma unroll
      for (int r = 0; r < 8; ++r) {
        int nG = w * 32 + i * 16 + (lane >> 4) * 8 + r;
        int mG = mt * 16 + (lane & 15);
        atomicAdd(&base[nG * MFEA + mG], acc[i][mt][r]);
      }
}

// ----- small supporting kernels --------------------------------------------
__global__ __launch_bounds__(256) void cvt_weight(const float* src, _Float16* dst,
                                                  int K, int N, int Kpad, int doT) {
  int i = blockIdx.x * 256 + threadIdx.x;
  if (i >= N * Kpad) return;
  int n = i / Kpad, k = i % Kpad;
  float v = 0.f;
  if (k < K) v = doT ? src[k * N + n] : src[n * K + k];
  dst[i] = (_Float16)v;
}

__global__ __launch_bounds__(256) void cvtf16_k(const float* src, _Float16* dst, long n) {
  long i = (long)blockIdx.x * 256 + threadIdx.x;
  if (i < n) dst[i] = (_Float16)src[i];
}

__global__ __launch_bounds__(256) void unfold_ln_k(const float* x, const float* g1,
                                                   const float* b1, _Float16* h16) {
  long tk = (long)blockIdx.x * 256 + threadIdx.x;
  if (tk >= TOK) return;
  int b = (int)(tk >> 14), pos = (int)(tk & 16383);
  int hh = pos >> 7, ww = pos & 127;
  float vals[48];
  int idx = 0; float s = 0.f;
  for (int c = 0; c < 3; ++c)
    for (int kh = 0; kh < 4; ++kh)
      for (int kw = 0; kw < 4; ++kw) {
        int r = hh + kh - 1, q = ww + kw - 1;
        float v = 0.f;
        if (r >= 0 && r < 129 && q >= 0 && q < 129)
          v = x[(((long)b * 3 + c) * 129 + r) * 129 + q];
        vals[idx++] = v; s += v;
      }
  float mu = s * (1.f / 48.f);
  float var = 0.f;
  for (int i = 0; i < 48; ++i) { float d = vals[i] - mu; var += d * d; }
  float rinv = rsqrtf(var * (1.f / 48.f) + 1e-5f);
  _Float16* o = h16 + tk * 64;
  for (int i = 0; i < 48; ++i) o[i] = (_Float16)((vals[i] - mu) * rinv * g1[i] + b1[i]);
  for (int i = 48; i < 64; ++i) o[i] = (_Float16)0.f;
}

__global__ __launch_bounds__(256) void xd_k(const _Float16* k16, const _Float16* q16,
                                            float* xdk, float* xdq) {
  long t = (long)blockIdx.x * 256 + threadIdx.x;
  float sk = 0.f, sq = 0.f;
  for (int i = 0; i < 256; ++i) {
    float a = (float)k16[t * 256 + i]; sk += a * a;
    float c = (float)q16[t * 256 + i]; sq += c * c;
  }
  xdk[t] = 0.5f * sk; xdq[t] = 0.5f * sq;
}

__global__ __launch_bounds__(128) void ksum_k(const _Float16* kp, float* ksum) {
  int b = blockIdx.y, ch = blockIdx.x, m = threadIdx.x;
  long tb = (long)b * TPB + (long)ch * 512;
  float s = 0.f;
  for (int i = 0; i < 512; ++i) s += (float)kp[(tb + i) * 128 + m];
  atomicAdd(&ksum[b * 128 + m], s);
}

__global__ __launch_bounds__(256) void d_k(const _Float16* qp, const float* ksum,
                                           float* D) {
  long t = (long)blockIdx.x * 256 + threadIdx.x;
  int b = (int)(t >> 14);
  float s = 0.f;
  for (int m = 0; m < 128; ++m) s += (float)qp[t * 128 + m] * ksum[b * 128 + m];
  D[t] = s;
}

__global__ __launch_bounds__(256) void ln2_k(const float* y, const float* g,
                                             const float* b, _Float16* z) {
  __shared__ float red[256];
  long t = blockIdx.x;
  int i = threadIdx.x;
  float v = y[t * 256 + i];
  red[i] = v; __syncthreads();
  for (int s = 128; s > 0; s >>= 1) { if (i < s) red[i] += red[i + s]; __syncthreads(); }
  float mu = red[0] * (1.f / 256.f); __syncthreads();
  float d = v - mu;
  red[i] = d * d; __syncthreads();
  for (int s = 128; s > 0; s >>= 1) { if (i < s) red[i] += red[i + s]; __syncthreads(); }
  float rinv = rsqrtf(red[0] * (1.f / 256.f) + 1e-5f);
  z[t * 256 + i] = (_Float16)(d * rinv * g[i] + b[i]);
}

// ----- 2x bilinear upsample, align_corners=True (float4 stores) -------------
__global__ __launch_bounds__(256) void upsample_k(const float* y, float* out) {
  long o4 = ((long)blockIdx.x * 256 + threadIdx.x) * 4;  // [B,256,256,256]
  int wo0 = (int)(o4 & 255);
  int ho  = (int)((o4 >> 8) & 255);
  int c   = (int)((o4 >> 16) & 255);
  int b   = (int)(o4 >> 24);
  float fh = ho * (127.0f / 255.0f);
  int h0 = (int)fh; if (h0 > 126) h0 = 126; float th = fh - (float)h0;
  const float* base = y + ((long)b * TPB) * 256 + c;
  float4 r;
  float* rp = &r.x;
#pragma unroll
  for (int j = 0; j < 4; ++j) {
    int wo = wo0 + j;
    float fw = wo * (127.0f / 255.0f);
    int w0 = (int)fw; if (w0 > 126) w0 = 126; float tw = fw - (float)w0;
    const float* p00 = base + ((long)h0 * 128 + w0) * 256;
    float a0 = p00[0]         * (1.f - tw) + p00[256]         * tw;
    float a1 = p00[128 * 256] * (1.f - tw) + p00[128 * 256 + 256] * tw;
    rp[j] = a0 * (1.f - th) + a1 * th;
  }
  *(float4*)(out + o4) = r;
}

// ---------------------------------------------------------------------------
extern "C" void kernel_launch(void* const* d_in, const int* in_sizes, int n_in,
                              void* d_out, int out_size, void* d_ws, size_t ws_size,
                              hipStream_t stream) {
  (void)in_sizes; (void)n_in; (void)out_size; (void)ws_size;
  const float* x      = (const float*)d_in[0];
  const float* g1     = (const float*)d_in[1];
  const float* b1     = (const float*)d_in[2];
  const float* w_kqv  = (const float*)d_in[3];
  const float* b_kqv  = (const float*)d_in[4];
  const float* w_prm  = (const float*)d_in[5];
  const float* w_proj = (const float*)d_in[6];
  const float* b_proj = (const float*)d_in[7];
  const float* g2     = (const float*)d_in[8];
  const float* b2     = (const float*)d_in[9];
  const float* w_mlp1 = (const float*)d_in[10];
  const float* b_mlp1 = (const float*)d_in[11];
  const float* w_mlp2 = (const float*)d_in[12];
  const float* b_mlp2 = (const float*)d_in[13];

  char* ws = (char*)d_ws;
  _Float16* h16    = (_Float16*)(ws + OFF_H16);
  _Float16* k16    = (_Float16*)(ws + OFF_K16);
  _Float16* q16    = (_Float16*)(ws + OFF_Q16);
  float*    v32    = (float*)   (ws + OFF_V32);
  _Float16* kp16   = (_Float16*)(ws + OFF_KP16);
  _Float16* qp16   = (_Float16*)(ws + OFF_QP16);
  float*    y32    = (float*)   (ws + OFF_Y32);
  float*    xdk    = (float*)   (ws + OFF_XDK);
  float*    xdq    = (float*)   (ws + OFF_XDQ);
  float*    Dv     = (float*)   (ws + OFF_DV);
  float*    ksumP  = (float*)   (ws + OFF_KSUM);
  float*    kptv32 = (float*)   (ws + OFF_KPTV32);
  _Float16* kptv16 = (_Float16*)(ws + OFF_KPTV16);
  _Float16* wkqvT  = (_Float16*)(ws + OFF_WKQVT);
  _Float16* wprm16 = (_Float16*)(ws + OFF_WPRM);
  _Float16* wprojT = (_Float16*)(ws + OFF_WPROJ);
  _Float16* wm1T   = (_Float16*)(ws + OFF_WM1);
  _Float16* wm2T   = (_Float16*)(ws + OFF_WM2);
  // buffer reuse (lifetimes verified by kernel order):
  _Float16* yattn16 = k16;                  // k16 dead after xd + kp gemm
  _Float16* z16     = q16;                  // q16 dead after qp gemm
  _Float16* h1      = kp16;                 // kp/qp dead after kptv/ksum/D/yattn
  float*    yfin    = v32;                  // v32 dead after proj skip

  // zero the atomically-accumulated buffers (ksum + kptv32 are contiguous)
  (void)hipMemsetAsync(ws + OFF_KSUM, 0, 4ull * MFEA * 4 + 4ull * EMB * MFEA * 4, stream);

  // weight prep: f16 convert (+ transpose into Bt[n][k] layout)
  cvt_weight<<<(KQV * KPAD + 255) / 256, 256, 0, stream>>>(w_kqv, wkqvT, 48, KQV, KPAD, 1);
  cvt_weight<<<(MFEA * EMB + 255) / 256, 256, 0, stream>>>(w_prm, wprm16, EMB, MFEA, EMB, 0);
  cvt_weight<<<(EMB * EMB + 255) / 256, 256, 0, stream>>>(w_proj, wprojT, EMB, EMB, EMB, 1);
  cvt_weight<<<(EMB * EMB + 255) / 256, 256, 0, stream>>>(w_mlp1, wm1T, EMB, EMB, EMB, 1);
  cvt_weight<<<(EMB * EMB + 255) / 256, 256, 0, stream>>>(w_mlp2, wm2T, EMB, EMB, EMB, 1);

  // 1) unfold + LN1 -> h16 [TOK,64]
  unfold_ln_k<<<TOK / 256, 256, 0, stream>>>(x, g1, b1, h16);

  // 2) kqv GEMM: [TOK,64] x [768,64]^T -> k16,q16 (f16), v32 (f32)
  { GP p{}; p.A = h16; p.ldA = KPAD; p.Bt = wkqvT; p.ldB = KPAD; p.K = KPAD;
    p.bias = b_kqv; p.o16 = k16; p.o16b = q16; p.o32 = v32; p.ldOut = EMB;
    gemm_tok<0><<<dim3(KQV / 64, TOK / 128), 256, 0, stream>>>(p); }

  // 3) xd = 0.5*|k|^2 / 0.5*|q|^2
  xd_k<<<TOK / 256, 256, 0, stream>>>(k16, q16, xdk, xdq);

  // 4) performer features kp, qp: exp(k@w_prm^T - xd)/sqrt(M)
  { GP p{}; p.A = k16; p.ldA = EMB; p.Bt = wprm16; p.ldB = EMB; p.K = EMB;
    p.xd = xdk; p.o16 = kp16; p.ldOut = MFEA;
    gemm_tok<1><<<dim3(MFEA / 64, TOK / 128), 256, 0, stream>>>(p);
    p.A = q16; p.xd = xdq; p.o16 = qp16;
    gemm_tok<1><<<dim3(MFEA / 64, TOK / 128), 256, 0, stream>>>(p); }

  // 5) kptv[b,n,m] = sum_t v*kp  (WMMA + global f32 atomics), ksum, D
  kptv_k<<<dim3(32, 4), 256, 0, stream>>>(v32, kp16, kptv32);
  ksum_k<<<dim3(32, 4), 128, 0, stream>>>(kp16, ksumP);
  cvtf16_k<<<(4 * EMB * MFEA + 255) / 256, 256, 0, stream>>>(kptv32, kptv16, 4l * EMB * MFEA);
  d_k<<<TOK / 256, 256, 0, stream>>>(qp16, ksumP, Dv);

  // 6) attention: y = (qp @ kptv^T) / (D+eps) -> yattn16
  { GP p{}; p.A = qp16; p.ldA = MFEA; p.Bt = kptv16; p.ldB = MFEA; p.K = MFEA;
    p.btBatchStride = (long)EMB * MFEA; p.D = Dv; p.o16 = yattn16; p.ldOut = EMB;
    gemm_tok<2><<<dim3(EMB / 64, TOK / 128), 256, 0, stream>>>(p); }

  // 7) proj + v skip -> y32
  { GP p{}; p.A = yattn16; p.ldA = EMB; p.Bt = wprojT; p.ldB = EMB; p.K = EMB;
    p.bias = b_proj; p.skip = v32; p.o32 = y32; p.ldOut = EMB;
    gemm_tok<3><<<dim3(EMB / 64, TOK / 128), 256, 0, stream>>>(p); }

  // 8) LN2 -> z16
  ln2_k<<<TOK, 256, 0, stream>>>(y32, g2, b2, z16);

  // 9) mlp1 + exact gelu -> h1
  { GP p{}; p.A = z16; p.ldA = EMB; p.Bt = wm1T; p.ldB = EMB; p.K = EMB;
    p.bias = b_mlp1; p.o16 = h1; p.ldOut = EMB;
    gemm_tok<4><<<dim3(EMB / 64, TOK / 128), 256, 0, stream>>>(p); }

  // 10) mlp2 + y skip -> yfin
  { GP p{}; p.A = h1; p.ldA = EMB; p.Bt = wm2T; p.ldB = EMB; p.K = EMB;
    p.bias = b_mlp2; p.skip = y32; p.o32 = yfin; p.ldOut = EMB;
    gemm_tok<5><<<dim3(EMB / 64, TOK / 128), 256, 0, stream>>>(p); }

  // 11) 2x bilinear upsample (align_corners) -> d_out [4,256,256,256] f32
  upsample_k<<<(4l * 256 * 256 * 256) / 4 / 256, 256, 0, stream>>>(yfin, (float*)d_out);
}
